// EdgeConv_56676388438533
// MI455X (gfx1250) — matrix operand order repaired and checked by reading, not compiled, not run on hardware
//
#include <hip/hip_runtime.h>
#include <hip/hip_bf16.h>
#include <math.h>

// Problem constants (from reference): B=4, C=64, N=8192, K=16, COUT=64
#define BATCH 4
#define CIN   64
#define NPT   8192
#define KNB   16
#define COUT  64
#define EPS   1e-5f

#define NPOINTS (BATCH * NPT)          // 32768
#define WAVES_PER_BLK 4                // 128-thread blocks
#define PPW 16                         // points per wave (one center-tile group)
#define EDGE_BLOCKS (NPOINTS / (WAVES_PER_BLK * PPW))   // 512

typedef __attribute__((ext_vector_type(16))) __bf16 v16bf;
typedef __attribute__((ext_vector_type(8)))  float  v8f;

// ---------------------------------------------------------------------------
// 1) Transpose x (B,C,N) -> xt (B,N,C) so neighbor gathers are contiguous.
// ---------------------------------------------------------------------------
__global__ __launch_bounds__(256) void k_transpose(const float* __restrict__ x,
                                                   float* __restrict__ xt) {
  __shared__ float tile[32][33];
  const int b  = blockIdx.z;
  const int c0 = blockIdx.y * 32;
  const int n0 = blockIdx.x * 32;
  const int tx = threadIdx.x;          // 0..31
  const int ty = threadIdx.y;          // 0..7
  #pragma unroll
  for (int r = 0; r < 32; r += 8)
    tile[ty + r][tx] = x[(size_t)(b * CIN + c0 + ty + r) * NPT + n0 + tx];
  __syncthreads();
  #pragma unroll
  for (int r = 0; r < 32; r += 8)
    xt[((size_t)b * NPT + n0 + ty + r) * CIN + c0 + tx] = tile[tx][ty + r];
}

// ---------------------------------------------------------------------------
// 2) KNN: one thread per query point; candidates tiled through LDS.
//    Register-resident top-16 (self excluded), expanded-form d2 like the ref.
// ---------------------------------------------------------------------------
__global__ __launch_bounds__(256) void k_knn(const float* __restrict__ coords,
                                             int* __restrict__ idx) {
  __shared__ float scx[256], scy[256], scz[256], ssq[256];
  const int b   = blockIdx.x >> 5;               // 32 blocks per batch
  const int n   = ((blockIdx.x & 31) << 8) + threadIdx.x;
  const int tid = threadIdx.x;
  const float* cb = coords + (size_t)b * 3 * NPT;

  const float qx = cb[n];
  const float qy = cb[NPT + n];
  const float qz = cb[2 * NPT + n];
  const float qs = qx * qx + qy * qy + qz * qz;

  float dk[KNB];
  int   id[KNB];
  #pragma unroll
  for (int s = 0; s < KNB; ++s) { dk[s] = INFINITY; id[s] = 0; }

  for (int tile = 0; tile < NPT / 256; ++tile) {
    const int j = (tile << 8) + tid;
    const float cx = cb[j], cy = cb[NPT + j], cz = cb[2 * NPT + j];
    __syncthreads();                    // previous tile fully consumed
    scx[tid] = cx; scy[tid] = cy; scz[tid] = cz;
    ssq[tid] = cx * cx + cy * cy + cz * cz;
    __syncthreads();
    for (int u = 0; u < 256; ++u) {
      const int jj = (tile << 8) + u;
      const float dot = qx * scx[u] + qy * scy[u] + qz * scz[u];
      const float d2  = qs + ssq[u] - 2.0f * dot;
      if (jj != n && d2 < dk[KNB - 1]) {
        float cd = d2; int ci = jj;
        #pragma unroll
        for (int s = 0; s < KNB; ++s) {     // unrolled insertion, stays in VGPRs
          const bool sw = cd < dk[s];
          const float td = dk[s]; const int ti = id[s];
          dk[s] = sw ? cd : td;  id[s] = sw ? ci : ti;
          cd    = sw ? td : cd;  ci    = sw ? ti : ci;
        }
      }
    }
  }
  int* op = idx + ((size_t)b * NPT + n) * KNB;
  #pragma unroll
  for (int s = 0; s < KNB; ++s) op[s] = id[s];
}

// ---------------------------------------------------------------------------
// A-fragment loader: 16-bit A 16x32 layout. For lane half `hi`, the fragment
// is two contiguous 8-element (16 B) runs: K = hi*16+(0..7) and hi*16+8+(0..7)
// within the 32-wide K-step window -> compiler emits 2x ds_load_b128.
// ---------------------------------------------------------------------------
__device__ __forceinline__ v16bf load_afrag(const __bf16* __restrict__ row,
                                            int kk, int hi) {
  const __bf16* p0 = row + kk * 32 + hi * 16;
  const __bf16* p1 = p0 + 8;
  v16bf a;
  #pragma unroll
  for (int j = 0; j < 8; ++j) { a[j] = p0[j]; a[8 + j] = p1[j]; }
  return a;
}

// ---------------------------------------------------------------------------
// 3/5) EdgeConv GEMM via v_wmma_f32_16x16x32_bf16, center/diff decomposition:
//   h[n,k,:] = center_n @ W_top + (nb - center_n) @ W_bot
//   Per wave: one 16-point group. Center pass = one 16(pts)x64x64 GEMM
//   (8 wmma), result cached as f32 16x64 in LDS. Per point: 16(nbrs)x64x64
//   GEMM (8 wmma) against register-resident W_bot fragments + broadcast add.
//   All 16 W fragments live in VGPRs (loaded once from transposed LDS W).
//   STATS=true : per-channel sum/sumsq of h -> fixed-order block partial.
//   STATS=false: BN scale/shift + ReLU + max over 16 rows -> out (B,COUT,N).
//   Bias b cancels in BN (h - mean_h == g - mean_g), so it is folded away.
// ---------------------------------------------------------------------------
template <bool STATS>
__global__ __launch_bounds__(128) void k_edge(const float* __restrict__ xt,
                                              const int* __restrict__ idx,
                                              const float* __restrict__ W,
                                              const float* __restrict__ st,
                                              float* __restrict__ out_or_part) {
  // LDS: [0,16K) W transposed bf16 [col][K]; per wave 6KB region:
  //   bytes [0,2K): center tile bf16 (16 pts x 64 ch)   -- dead after ctr GEMM
  //   bytes [0,4K): center D-tile f32 (16 pts x 64 cols) -- overlays the above
  //   bytes [4K,6K): diff tile bf16 (16 nbrs x 64 ch)
  // then 2KB stats partials.
  __shared__ __align__(16) char smem[16 * 1024 + WAVES_PER_BLK * 6 * 1024 + 2048];

  const int tid  = threadIdx.x;
  const int lane = tid & 31;
  const int wave = tid >> 5;
  const int hi   = lane >> 4;     // which half of the wave (0/1)
  const int m    = lane & 15;     // row (A/C) or column (B/C) index

  __bf16* sWt  = (__bf16*)smem;                              // 64 x 128
  char*   wb   = smem + 16 * 1024 + wave * 6 * 1024;
  __bf16* sCb  = (__bf16*)wb;                                // 16 x 64 bf16
  float*  sCt  = (float*)wb;                                 // 16 x 64 f32
  __bf16* sD   = (__bf16*)(wb + 4 * 1024);                   // 16 x 64 bf16
  float*  sRed = (float*)(smem + 16 * 1024 + WAVES_PER_BLK * 6 * 1024);

  // ---- stage W transposed: sWt[col*128 + K] = W[K*64 + col] ----
  for (int i = tid; i < 128 * 64; i += 128) {
    const int col = i >> 7, kg = i & 127;
    sWt[i] = (__bf16)W[kg * 64 + col];
  }
  __syncthreads();

  // ---- hoist all 16 W fragments into VGPRs (2x ds_load_b128 each) ----
  // wf[h2][kk][t]: h2 selects W_top/W_bot, kk the 32-wide K-step, t the
  // 16-wide column tile. B 32x16, in this code's k order (matching A): lane m = column, lanes 0-15 hold
  // K 0..15, lanes 16-31 hold K 16..31 (contiguous in transposed W).
  v16bf wf[2][2][4];
  #pragma unroll
  for (int h2 = 0; h2 < 2; ++h2)
    #pragma unroll
    for (int kk = 0; kk < 2; ++kk)
      #pragma unroll
      for (int t = 0; t < 4; ++t) {
        const __bf16* p = sWt + (t * 16 + m) * 128 + h2 * 64 + kk * 32 + hi * 16;
        v16bf w;
        #pragma unroll
        for (int j = 0; j < 16; ++j) w[j] = p[j];
        wf[h2][kk][t] = w;
      }

  float scale[4], shift[4];
  if (!STATS) {
    #pragma unroll
    for (int t = 0; t < 4; ++t) {
      const int col = t * 16 + m;
      scale[t] = st[col];
      shift[t] = st[64 + col];
    }
  }

  float cs[4] = {0.f, 0.f, 0.f, 0.f};
  float cq[4] = {0.f, 0.f, 0.f, 0.f};

  const int base = (blockIdx.x * WAVES_PER_BLK + wave) * PPW;  // 16 points
  const int b    = base >> 13;
  const int c0   = 2 * lane;                                   // 2 ch / lane

  // ---- stage center tile: rows = the wave's 16 points ----
  for (int pt = 0; pt < PPW; ++pt) {
    const float2 cen = ((const float2*)(xt + (size_t)(base + pt) * CIN))[lane];
    sCb[pt * 64 + c0]     = (__bf16)cen.x;
    sCb[pt * 64 + c0 + 1] = (__bf16)cen.y;
  }

  // ---- center GEMM: 16 pts x 64 x 64 -> ctile, spill f32 to LDS ----
  {
    v8f ct[4];
    #pragma unroll
    for (int t = 0; t < 4; ++t)
      #pragma unroll
      for (int r = 0; r < 8; ++r) ct[t][r] = 0.f;
    #pragma unroll
    for (int kk = 0; kk < 2; ++kk) {
      const v16bf a = load_afrag(sCb + m * 64, kk, hi);
      #pragma unroll
      for (int t = 0; t < 4; ++t)
        ct[t] = __builtin_amdgcn_wmma_f32_16x16x32_bf16(
            false, a, false, wf[0][kk][t], (short)0, ct[t], false, false);
    }
    // D layout: lane = col t*16+m, VGPR r = row r+8*hi. Overlay onto sCb.
    #pragma unroll
    for (int t = 0; t < 4; ++t)
      #pragma unroll
      for (int r = 0; r < 8; ++r)
        sCt[(r + 8 * hi) * 64 + t * 16 + m] = ct[t][r];
  }

  // ---- per point: diff GEMM + broadcast center + epilogue ----
  for (int pt = 0; pt < PPW; ++pt) {
    const int p = base + pt;
    const int n = p & (NPT - 1);

    const float2 cen = ((const float2*)(xt + (size_t)p * CIN))[lane];
    const int* ip = idx + (size_t)p * KNB;
    for (int k = 0; k < KNB; ++k) {
      const int j = ip[k];                                  // wave-uniform
      const float2 nb =
          ((const float2*)(xt + ((size_t)(b << 13) + j) * CIN))[lane];
      sD[k * 64 + c0]     = (__bf16)(nb.x - cen.x);
      sD[k * 64 + c0 + 1] = (__bf16)(nb.y - cen.y);
    }

    v8f acc[4];
    #pragma unroll
    for (int t = 0; t < 4; ++t)
      #pragma unroll
      for (int r = 0; r < 8; ++r) acc[t][r] = 0.f;
    #pragma unroll
    for (int kk = 0; kk < 2; ++kk) {
      const v16bf a = load_afrag(sD + m * 64, kk, hi);
      #pragma unroll
      for (int t = 0; t < 4; ++t)
        acc[t] = __builtin_amdgcn_wmma_f32_16x16x32_bf16(
            false, a, false, wf[1][kk][t], (short)0, acc[t], false, false);
    }

    if (STATS) {
      #pragma unroll
      for (int t = 0; t < 4; ++t) {
        const float cval = sCt[pt * 64 + t * 16 + m];   // center row broadcast
        #pragma unroll
        for (int r = 0; r < 8; ++r) {
          const float v = acc[t][r] + cval;
          cs[t] += v;
          cq[t] += v * v;
        }
      }
    } else {
      #pragma unroll
      for (int t = 0; t < 4; ++t) {
        const float cval = sCt[pt * 64 + t * 16 + m];
        float mx = 0.f;                        // ReLU output is >= 0
        #pragma unroll
        for (int r = 0; r < 8; ++r) {
          float v = fmaf(acc[t][r] + cval, scale[t], shift[t]);
          v = v > 0.f ? v : 0.f;
          mx = fmaxf(mx, v);
        }
        mx = fmaxf(mx, __shfl_xor(mx, 16, 32));   // combine rows 0-7 / 8-15
        if (hi == 0) {
          const int col = t * 16 + m;
          out_or_part[((size_t)(b * COUT + col) << 13) + n] = mx;
        }
      }
    }
  }

  if (STATS) {
    #pragma unroll
    for (int t = 0; t < 4; ++t) {
      cs[t] += __shfl_xor(cs[t], 16, 32);
      cq[t] += __shfl_xor(cq[t], 16, 32);
    }
    if (hi == 0) {
      #pragma unroll
      for (int t = 0; t < 4; ++t) {
        sRed[wave * 128 + t * 16 + m]      = cs[t];
        sRed[wave * 128 + 64 + t * 16 + m] = cq[t];
      }
    }
    __syncthreads();
    if (tid < 128) {                          // fixed-order => deterministic
      float s = 0.f;
      #pragma unroll
      for (int w = 0; w < WAVES_PER_BLK; ++w) s += sRed[w * 128 + tid];
      out_or_part[(size_t)blockIdx.x * 128 + tid] = s;
    }
  }
}

// ---------------------------------------------------------------------------
// 4) Deterministic reduction of block partials -> per-channel scale/shift.
//    h = g + bias; BN(h) = (g - mean_g)*inv*gamma + beta  (bias cancels).
// ---------------------------------------------------------------------------
__global__ void k_finalize(const float* __restrict__ part,
                           const float* __restrict__ gamma,
                           const float* __restrict__ beta,
                           float* __restrict__ st) {
  const int c = threadIdx.x;                 // 0..63
  float s = 0.f, q = 0.f;
  for (int blk = 0; blk < EDGE_BLOCKS; ++blk) {
    s += part[(size_t)blk * 128 + c];
    q += part[(size_t)blk * 128 + 64 + c];
  }
  const float inv  = 1.0f / (float)(NPOINTS * KNB);   // 1/524288
  const float mean = s * inv;
  const float var  = q * inv - mean * mean;
  const float sc   = gamma[c] * rsqrtf(var + EPS);
  st[c]      = sc;
  st[64 + c] = beta[c] - mean * sc;
}

// ---------------------------------------------------------------------------
// Launch. Workspace layout (bytes):
//   [0, 8388608)           xt      : B*N*C f32 transposed features
//   [8388608, 10485760)    idx     : B*N*16 i32 neighbor indices
//   [10485760, 10747904)   part    : 512 * 128 f32 BN stat partials
//   [10747904, 10748416)   st      : 128 f32 (scale[64] | shift[64])
// ---------------------------------------------------------------------------
extern "C" void kernel_launch(void* const* d_in, const int* in_sizes, int n_in,
                              void* d_out, int out_size, void* d_ws, size_t ws_size,
                              hipStream_t stream) {
  const float* x      = (const float*)d_in[0];
  const float* coords = (const float*)d_in[1];
  const float* W      = (const float*)d_in[2];
  // d_in[3] = bias: cancels inside BatchNorm, unused.
  const float* gamma  = (const float*)d_in[4];
  const float* beta   = (const float*)d_in[5];
  float* out = (float*)d_out;

  char* ws   = (char*)d_ws;
  float* xt  = (float*)(ws);
  int*   idx = (int*)(ws + 8388608);
  float* prt = (float*)(ws + 10485760);
  float* st  = (float*)(ws + 10747904);

  dim3 tb(32, 8);
  dim3 tg(NPT / 32, CIN / 32, BATCH);
  k_transpose<<<tg, tb, 0, stream>>>(x, xt);

  k_knn<<<BATCH * (NPT / 256), 256, 0, stream>>>(coords, idx);

  k_edge<true><<<EDGE_BLOCKS, 128, 0, stream>>>(xt, idx, W, nullptr, prt);
  k_finalize<<<1, COUT, 0, stream>>>(prt, gamma, beta, st);
  k_edge<false><<<EDGE_BLOCKS, 128, 0, stream>>>(xt, idx, W, st, out);
}
